// Context_UpsampleLayer_6047313953089
// MI455X (gfx1250) — compile-verified, matrix operand order checked by reading, not run on hardware
//
#include <hip/hip_runtime.h>
#include <cstdint>
#include <cstddef>

typedef __attribute__((ext_vector_type(16))) _Float16 v16h;
typedef __attribute__((ext_vector_type(8)))  float    v8f;

#define N_IN 60000
#define N_UP 480000

union Frag16 {
  v16h v;
  uint4 q[2];
};

__device__ __forceinline__ uint4 zero4() { uint4 z; z.x = 0u; z.y = 0u; z.z = 0u; z.w = 0u; return z; }

// ---------------------------------------------------------------------------
// Repack f32 weights [kvol][cin][cout] into WMMA B-fragment layout (f16):
// dst[(((k*coutt+ct)*cinch+cc)*32 + lane)*16 + s], where for 16-bit 32x16 B:
//   lane<16 : K = (s<8 ? s : s+8),  lane>=16 : K = (s<8 ? s+8 : s+16); N = lane&15
// Zero-pads K>=cin and N>=cout.
// ---------------------------------------------------------------------------
__global__ __launch_bounds__(256) void repack_w(const float* __restrict__ W, _Float16* __restrict__ dst,
                                                int kvol, int cin, int cout, int cinch, int coutt) {
  int e = blockIdx.x * 256 + threadIdx.x;
  int total = kvol * coutt * cinch * 512;
  if (e >= total) return;
  int s    = e & 15;
  int lane = (e >> 4) & 31;
  int rem  = e >> 9;
  int cc = rem % cinch; rem /= cinch;
  int ct = rem % coutt;
  int k  = rem / coutt;
  int kslot = (lane < 16) ? ((s < 8) ? s : s + 8) : ((s < 8) ? s + 8 : s + 16);
  int K = cc * 32 + kslot;
  int N = ct * 16 + (lane & 15);
  float v = (K < cin && N < cout) ? W[((size_t)k * cin + K) * cout + N] : 0.f;
  dst[e] = (_Float16)v;
}

// ---------------------------------------------------------------------------
// Build concat(x_feats[64], ctx_feats[8]) padded to 96 cols, converted to f16.
// ---------------------------------------------------------------------------
__global__ __launch_bounds__(256) void build_xc(const float* __restrict__ x, const float* __restrict__ c,
                                                _Float16* __restrict__ out, int n) {
  int i = blockIdx.x * 256 + threadIdx.x;
  if (i >= n * 96) return;
  int r = i / 96, col = i % 96;
  float v = (col < 64) ? x[(size_t)r * 64 + col]
                       : ((col < 72) ? c[(size_t)r * 8 + (col - 64)] : 0.f);
  out[i] = (_Float16)v;
}

__global__ __launch_bounds__(256) void cvt_f32_f16(const float* __restrict__ in, _Float16* __restrict__ out, int n) {
  int i = blockIdx.x * 256 + threadIdx.x;
  if (i < n) out[i] = (_Float16)in[i];
}

// ---------------------------------------------------------------------------
// Generic WMMA sparse conv.
//   - 8 wave32s / block; each wave owns MT consecutive 16-row output tiles
//     (register M-blocking: each B fragment feeds MT WMMAs).
//   - Per tap k: per-lane gather into the 16-bit A-fragment layout
//     (lanes 0-15: K {0..7,16..23} of row M=lane; lanes 16-31: K {8..15,24..31}),
//     rows with nbr==-1 zeroed by select (EXEC stays all-ones for WMMA).
//   - STAGE: whole repacked weight block staged to LDS once per workgroup
//     (cooperative b128 copy + barrier), B fragments then come from ds_load.
//   - nbr==nullptr -> identity gather (dense per-point matmul).
// ---------------------------------------------------------------------------
template<int CIN, int COUT, int COUTPAD, int COUTSTORE, int MT, bool RELU, bool ADDTO, bool STAGE>
__global__ __launch_bounds__(256) void conv_wmma(
    const _Float16* __restrict__ feats,
    const int* __restrict__ nbr, int kvol, int nOut,
    const _Float16* __restrict__ wfrag, const float* __restrict__ bias,
    float* __restrict__ outF, _Float16* __restrict__ outB,
    int outStride, int outOffset) {
  constexpr int CINCH   = (CIN + 31) / 32;
  constexpr int FSTRIDE = CINCH * 32;            // padded f16 row stride of feats
  constexpr int COUTT   = COUTPAD / 16;          // stored column tiles
  constexpr int ACTIVE  = (COUT + 15) / 16;      // column tiles actually computed

  extern __shared__ _Float16 smw[];
  if (STAGE) {
    const int tot4 = (kvol * COUTT * CINCH * 512) >> 3;  // uint4 chunks
    const uint4* gsrc = (const uint4*)wfrag;
    uint4* ldst = (uint4*)smw;
    for (int i = threadIdx.x; i < tot4; i += 256) ldst[i] = gsrc[i];
    __syncthreads();
  }
  const _Float16* wb = STAGE ? (const _Float16*)smw : wfrag;

  const int lane  = threadIdx.x & 31;
  const int wave  = threadIdx.x >> 5;
  const int tile0 = (blockIdx.x * 8 + wave) * MT;
  if (tile0 * 16 >= nOut) return;                // wave-uniform exit
  const int sub = lane & 15;
  const int hi  = (lane >> 4) & 1;

  v8f acc[MT][ACTIVE];
  v8f z8 = {0.f, 0.f, 0.f, 0.f, 0.f, 0.f, 0.f, 0.f};
#pragma unroll
  for (int mt = 0; mt < MT; ++mt)
#pragma unroll
    for (int ct = 0; ct < ACTIVE; ++ct) acc[mt][ct] = z8;

  for (int k = 0; k < kvol; ++k) {
    Frag16 a[MT][CINCH];
#pragma unroll
    for (int mt = 0; mt < MT; ++mt) {
      const int myRow = (tile0 + mt) * 16 + sub;
      const int idx = (myRow < nOut) ? (nbr ? nbr[(size_t)k * nOut + myRow] : myRow) : -1;
      const _Float16* src = feats + (size_t)(idx < 0 ? 0 : idx) * FSTRIDE;
#pragma unroll
      for (int cc = 0; cc < CINCH; ++cc) {
        const int off0 = cc * 32 + hi * 8;
        a[mt][cc].q[0] = *(const uint4*)(src + off0);
        a[mt][cc].q[1] = *(const uint4*)(src + off0 + 16);
        if (idx < 0) { a[mt][cc].q[0] = zero4(); a[mt][cc].q[1] = zero4(); }
      }
    }
#pragma unroll
    for (int ct = 0; ct < ACTIVE; ++ct) {
#pragma unroll
      for (int cc = 0; cc < CINCH; ++cc) {
        const _Float16* wp = wb + ((((size_t)k * COUTT + ct) * CINCH + cc) * 32 + lane) * 16;
        Frag16 b;
        b.q[0] = *(const uint4*)(wp);
        b.q[1] = *(const uint4*)(wp + 8);
#pragma unroll
        for (int mt = 0; mt < MT; ++mt) {
          acc[mt][ct] = __builtin_amdgcn_wmma_f32_16x16x32_f16(
              false, a[mt][cc].v, false, b.v, (short)0, acc[mt][ct], false, false);
        }
      }
    }
  }

  // C/D layout: VGPR r holds row rowBase + r + 8*hi, column = ct*16 + (lane&15)
#pragma unroll
  for (int mt = 0; mt < MT; ++mt) {
    const int rowB = (tile0 + mt) * 16;
#pragma unroll
    for (int ct = 0; ct < COUTT; ++ct) {
      const int col = ct * 16 + sub;
      if (col >= COUTSTORE) continue;
#pragma unroll
      for (int r = 0; r < 8; ++r) {
        const int row = rowB + r + hi * 8;
        if (row >= nOut) continue;
        float v = 0.f;
        if (ct < ACTIVE && col < COUT) {
          v = acc[mt][(ct < ACTIVE) ? ct : 0][r] + bias[col];
          if (RELU) v = fmaxf(v, 0.f);
        }
        const size_t o = (size_t)row * outStride + outOffset + col;
        if (outF) { if (ADDTO) outF[o] += v; else outF[o] = v; }
        if (outB) outB[o] = (_Float16)v;
      }
    }
  }
}

// ---------------------------------------------------------------------------
// keep / argmax-fallback machinery
// ---------------------------------------------------------------------------
__device__ __forceinline__ unsigned int fkey(float f) {
  unsigned int b = __float_as_uint(f);
  return (b & 0x80000000u) ? ~b : (b | 0x80000000u);   // order-preserving map
}

__global__ void init_red(unsigned long long* red, unsigned int* flag) {
  if (threadIdx.x == 0) { *red = 0ull; *flag = 0u; }
}

__global__ __launch_bounds__(256) void reduce_cls(const float* __restrict__ cls, int n,
                                                  unsigned long long* __restrict__ red,
                                                  unsigned int* __restrict__ flag) {
  __shared__ unsigned long long sm[256];
  unsigned long long best = 0ull;
  unsigned int any = 0u;
  for (int i = blockIdx.x * 256 + threadIdx.x; i < n; i += gridDim.x * 256) {
    float v = cls[i];
    any |= (v > 0.f) ? 1u : 0u;
    // high 32: ordered value key; low 32: ~i so ties pick the smallest index
    unsigned long long p = ((unsigned long long)fkey(v) << 32) | (unsigned long long)(~(unsigned int)i);
    best = (p > best) ? p : best;
  }
  sm[threadIdx.x] = best;
  if (any) atomicOr(flag, 1u);
  __syncthreads();
  for (int s = 128; s > 0; s >>= 1) {
    if ((int)threadIdx.x < s) {
      if (sm[threadIdx.x + s] > sm[threadIdx.x]) sm[threadIdx.x] = sm[threadIdx.x + s];
    }
    __syncthreads();
  }
  if (threadIdx.x == 0) atomicMax(red, sm[0]);
}

__global__ __launch_bounds__(256) void finalize_k(const float* __restrict__ hF, const float* __restrict__ cls,
                                                  const unsigned long long* __restrict__ red,
                                                  const unsigned int* __restrict__ flag,
                                                  float* __restrict__ outP, float* __restrict__ outK, int n) {
  int r = blockIdx.x * 256 + threadIdx.x;
  if (r >= n) return;
  unsigned long long p = *red;
  unsigned int fl = *flag;
  int arg = (int)(~(unsigned int)(p & 0xFFFFFFFFull));
  bool keep = (cls[r] > 0.f) || (fl == 0u && r == arg);
  outK[r] = keep ? 1.f : 0.f;
  float m = keep ? 1.f : 0.f;
#pragma unroll
  for (int c = 0; c < 32; ++c) outP[(size_t)r * 32 + c] = hF[(size_t)r * 32 + c] * m;
}

// ---------------------------------------------------------------------------
// Launch
// ---------------------------------------------------------------------------
extern "C" void kernel_launch(void* const* d_in, const int* in_sizes, int n_in,
                              void* d_out, int out_size, void* d_ws, size_t ws_size,
                              hipStream_t stream) {
  const float* x_feats = (const float*)d_in[0];
  const float* ctx     = (const float*)d_in[1];
  const float* W_ctx   = (const float*)d_in[2];
  const float* b_ctx   = (const float*)d_in[3];
  const float* W_up    = (const float*)d_in[4];
  const float* b_up    = (const float*)d_in[5];
  const float* W_conv  = (const float*)d_in[6];
  const float* b_conv  = (const float*)d_in[7];
  const float* w00     = (const float*)d_in[8];
  const float* b00     = (const float*)d_in[9];
  const float* w01     = (const float*)d_in[10];
  const float* b01     = (const float*)d_in[11];
  const float* w10     = (const float*)d_in[12];
  const float* b10     = (const float*)d_in[13];
  const float* w11     = (const float*)d_in[14];
  const float* b11     = (const float*)d_in[15];
  const float* w12     = (const float*)d_in[16];
  const float* b12     = (const float*)d_in[17];
  const float* W_cls   = (const float*)d_in[18];
  const float* b_cls   = (const float*)d_in[19];
  const int*   nbr_in  = (const int*)d_in[20];
  const int*   nbr_out = (const int*)d_in[21];

  char* wsb = (char*)d_ws;
  size_t off = 0;
  auto alloc = [&](size_t bytes) -> void* {
    void* p = wsb + off;
    off = (off + bytes + 255) & ~(size_t)255;
    return p;
  };

  _Float16* xc = (_Float16*)alloc((size_t)N_IN * 96 * 2);   // concat input, padded to 96
  _Float16* xf = (_Float16*)alloc((size_t)N_IN * 64 * 2);   // step-1 output (no relu)
  _Float16* up = (_Float16*)alloc((size_t)N_UP * 64 * 2);   // relu(up), [480000,64]
  float*    hF = (float*)   alloc((size_t)N_UP * 32 * 4);   // h, f32 master
  _Float16* hH = (_Float16*)alloc((size_t)N_UP * 32 * 2);   // h, f16 for gathers
  _Float16* tA = (_Float16*)alloc((size_t)N_UP * 32 * 2);   // temp (8ch padded to 32)
  _Float16* tB = (_Float16*)alloc((size_t)N_UP * 32 * 2);   // temp (8ch padded to 32)

  _Float16* fctx  = (_Float16*)alloc((size_t)27 * 4 * 3 * 512 * 2);
  _Float16* fup   = (_Float16*)alloc((size_t)8  * 4 * 2 * 512 * 2);
  _Float16* fconv = (_Float16*)alloc((size_t)27 * 2 * 2 * 512 * 2);
  _Float16* f00   = (_Float16*)alloc((size_t)81 * 2 * 1 * 512 * 2);
  _Float16* f01   = (_Float16*)alloc((size_t)81 * 1 * 1 * 512 * 2);
  _Float16* f10   = (_Float16*)alloc((size_t)3  * 2 * 1 * 512 * 2);
  _Float16* f11   = (_Float16*)alloc((size_t)81 * 2 * 1 * 512 * 2);
  _Float16* f12   = (_Float16*)alloc((size_t)3  * 1 * 1 * 512 * 2);
  _Float16* fcls  = (_Float16*)alloc((size_t)27 * 1 * 1 * 512 * 2);
  unsigned long long* red = (unsigned long long*)alloc(16);
  unsigned int* flag = (unsigned int*)(red + 1);

  auto rpk = [&](const float* W, _Float16* dst, int kvol, int cin, int cout, int cinch, int coutt) {
    int total = kvol * coutt * cinch * 512;
    repack_w<<<(total + 255) / 256, 256, 0, stream>>>(W, dst, kvol, cin, cout, cinch, coutt);
  };
  rpk(W_ctx,  fctx,  27, 72, 64, 3, 4);
  rpk(W_up,   fup,    8, 64, 64, 2, 4);
  rpk(W_conv, fconv, 27, 64, 32, 2, 2);
  rpk(w00,    f00,   81, 32,  8, 1, 2);
  rpk(w01,    f01,   81,  8, 16, 1, 1);
  rpk(w10,    f10,    3, 32,  8, 1, 2);
  rpk(w11,    f11,   81,  8,  8, 1, 2);
  rpk(w12,    f12,    3,  8, 16, 1, 1);
  rpk(W_cls,  fcls,  27, 32,  1, 1, 1);

  build_xc<<<(N_IN * 96 + 255) / 256, 256, 0, stream>>>(x_feats, ctx, xc, N_IN);

  auto cblk = [](int n, int mt) {
    int tiles = (n + 15) / 16;
    int waves = (tiles + mt - 1) / mt;
    return (waves + 7) / 8;
  };

  // LDS staging sizes (bytes) for the kvol=27 convs that fit in WGP LDS
  const int shConv = 27 * 2 * 2 * 512 * 2;  // 110592
  const int sh00   = 27 * 2 * 1 * 512 * 2;  // 55296
  const int sh01   = 27 * 1 * 1 * 512 * 2;  // 27648
  const int sh11   = 27 * 2 * 1 * 512 * 2;  // 55296
  const int shCls  = 27 * 1 * 1 * 512 * 2;  // 27648

  // 1) x = sparse_conv(concat, nbr_in, W_ctx) + b_ctx  (no relu; 324KB frags -> no LDS stage)
  conv_wmma<72, 64, 64, 64, 2, false, false, false><<<cblk(N_IN, 2), 256, 0, stream>>>(
      xc, nbr_in, 27, N_IN, fctx, b_ctx, nullptr, xf, 64, 0);

  // 2) up[n*8+child] = relu(x @ W_up[child] + b_up)
  for (int child = 0; child < 8; ++child) {
    conv_wmma<64, 64, 64, 64, 2, true, false, false><<<cblk(N_IN, 2), 256, 0, stream>>>(
        xf, nullptr, 1, N_IN, fup + (size_t)child * 4 * 2 * 512, b_up,
        nullptr, up, 512, child * 64);
  }

  // 3) h = relu(sparse_conv(up, nbr_out, W_conv) + b_conv)   (weights staged in LDS)
  conv_wmma<64, 32, 32, 32, 2, true, false, true><<<cblk(N_UP, 2), 256, shConv, stream>>>(
      up, nbr_out, 27, N_UP, fconv, b_conv, hF, hH, 32, 0);

  // 4) 3 InceptionResNet blocks; residual fused via ADDTO into hF
  for (int i = 0; i < 3; ++i) {
    conv_wmma<32, 8, 32, 32, 4, true, false, true><<<cblk(N_UP, 4), 256, sh00, stream>>>(
        hH, nbr_out, 27, N_UP, f00 + (size_t)i * 27 * 2 * 512, b00 + i * 8,
        nullptr, tA, 32, 0);
    conv_wmma<8, 16, 16, 16, 4, false, true, true><<<cblk(N_UP, 4), 256, sh01, stream>>>(
        tA, nbr_out, 27, N_UP, f01 + (size_t)i * 27 * 512, b01 + i * 16,
        hF, nullptr, 32, 0);
    conv_wmma<32, 8, 32, 32, 4, true, false, false><<<cblk(N_UP, 4), 256, 0, stream>>>(
        hH, nullptr, 1, N_UP, f10 + (size_t)i * 2 * 512, b10 + i * 8,
        nullptr, tA, 32, 0);
    conv_wmma<8, 8, 32, 32, 4, true, false, true><<<cblk(N_UP, 4), 256, sh11, stream>>>(
        tA, nbr_out, 27, N_UP, f11 + (size_t)i * 27 * 2 * 512, b11 + i * 8,
        nullptr, tB, 32, 0);
    conv_wmma<8, 16, 16, 16, 4, false, true, false><<<cblk(N_UP, 4), 256, 0, stream>>>(
        tB, nullptr, 1, N_UP, f12 + (size_t)i * 512, b12 + i * 16,
        hF, nullptr, 32, 16);
    cvt_f32_f16<<<(N_UP * 32 + 255) / 256, 256, 0, stream>>>(hF, hH, N_UP * 32);
  }

  // 5) cls = sparse_conv(h, nbr_out, W_cls)[:,0]  -> directly into d_out slot
  float* outP    = (float*)d_out;                  // [480000,32]
  float* clsOut  = outP + (size_t)N_UP * 32;       // [480000]
  float* keepOut = clsOut + N_UP;                  // [480000]
  conv_wmma<32, 1, 16, 1, 4, false, false, true><<<cblk(N_UP, 4), 256, shCls, stream>>>(
      hH, nbr_out, 27, N_UP, fcls, b_cls, clsOut, nullptr, 1, 0);

  // 6) keep = cls>0 (fallback: first argmax if all negative); out_pruned = h*keep
  init_red<<<1, 32, 0, stream>>>(red, flag);
  reduce_cls<<<480, 256, 0, stream>>>(clsOut, N_UP, red, flag);
  finalize_k<<<(N_UP + 255) / 256, 256, 0, stream>>>(hF, clsOut, red, flag, outP, keepOut, N_UP);
}
